// multi_head_attention_90915867722050
// MI455X (gfx1250) — matrix operand order repaired and checked by reading, not compiled, not run on hardware
//
#include <hip/hip_runtime.h>
#include <hip/hip_bf16.h>

typedef _Float16 f16;
typedef __attribute__((ext_vector_type(16))) _Float16 v16h;
typedef __attribute__((ext_vector_type(8)))  _Float16 f16x8;
typedef __attribute__((ext_vector_type(8)))  float    v8f;
typedef __attribute__((ext_vector_type(4)))  unsigned int v4u;
typedef __attribute__((ext_vector_type(8)))  int v8i;
typedef __attribute__((ext_vector_type(4)))  int v4i;

#define WMMA_F16(A,B,C) __builtin_amdgcn_wmma_f32_16x16x32_f16(false,(A),false,(B),(short)0,(C),false,false)

constexpr int BB = 16;    // batch
constexpr int SS = 511;   // seq
constexpr int SP = 512;   // padded seq
constexpr int DD = 1024;  // model dim
constexpr int HH = 16;    // heads
constexpr int DKV = 64;   // head dim

union H16 { v16h v; f16x8 h8[2]; };

// A-fragment (16x32, ISA dense-A layout): lane holds row L%16,
// halves i -> col cbase + (i%8) + 16*(i/8), cbase = kk + (L/16)*8.
__device__ __forceinline__ v16h frag_ld_a(const f16* rowptr, int cbase) {
  H16 u;
  u.h8[0] = *(const f16x8*)(rowptr + cbase);
  u.h8[1] = *(const f16x8*)(rowptr + cbase + 16);
  return u.v;
}
// B-fragment (32x16, dense-B layout): lane holds col L%16, kbase = kk + (L/16)*16,
// halves i -> K = kk + (L/16)*8 + (i%8) + 16*(i/8) (same as A).
__device__ __forceinline__ v16h frag_ld_b(const f16* rowptr, int kbase) {
  H16 u;
  const int kb8 = kbase - ((kbase >> 4) & 1) * 8;
  u.h8[0] = *(const f16x8*)(rowptr + kb8);
  u.h8[1] = *(const f16x8*)(rowptr + kb8 + 16);
  return u.v;
}

// ---- Tensor Data Mover: 2D f16 tile, tile == tensor, contiguous rows ----
// D# layout per CDNA5 ISA.
// This toolchain exposes the 6-arg builtin:
//   (uint32x4 g0, int32x8 g1, int32x4 g2, int32x4 g3, int32x8 extra, i32 cpol)
__device__ __forceinline__ void tdm_load_2d_f16(unsigned int lds_byte_off,
                                                const void* gptr,
                                                unsigned int dim0,   // row length (elems)
                                                unsigned int dim1) { // #rows
  const unsigned long long ga = (unsigned long long)(uintptr_t)gptr;
  v4u g0;
  g0[0] = 1u;                                        // count=1 (valid user D#)
  g0[1] = lds_byte_off;                              // lds_addr
  g0[2] = (unsigned int)(ga & 0xffffffffu);          // global_addr[31:0]
  g0[3] = (unsigned int)((ga >> 32) & 0x1ffffffu) | (2u << 30);  // addr[56:32] | type=2
  v8i g1;
  g1[0] = (int)(1u << 16);                           // wg_mask=0, data_size=1 (2B)
  g1[1] = (int)((dim0 & 0xffffu) << 16);             // tensor_dim0[15:0] at bits 63:48
  g1[2] = (int)(((dim0 >> 16) & 0xffffu) | ((dim1 & 0xffffu) << 16));   // dim0 hi | dim1 lo
  g1[3] = (int)(((dim1 >> 16) & 0xffffu) | ((dim0 & 0xffffu) << 16));   // dim1 hi | tile_dim0
  g1[4] = (int)(dim1 & 0xffffu);                     // tile_dim1 | tile_dim2=0
  g1[5] = (int)dim0;                                 // tensor_dim0_stride[31:0]
  g1[6] = 0;                                         // stride0 hi | stride1 lo
  g1[7] = 0;
  v4i gz4 = {0, 0, 0, 0};
  v8i gz8 = {0, 0, 0, 0, 0, 0, 0, 0};
  __builtin_amdgcn_tensor_load_to_lds(g0, g1, gz4, gz4, gz8, 0);
}

// ---------------- pack / convert kernels ----------------

__global__ void pack_x_kernel(const float* __restrict__ q, f16* __restrict__ Xt) {
  size_t i = (size_t)blockIdx.x * 256 + threadIdx.x;   // over B*S*D
  if (i >= (size_t)BB * SS * DD) return;
  int d = (int)(i & (DD - 1));
  size_t t = i >> 10;                 // b*S + s
  int s = (int)(t % SS);
  int b = (int)(t / SS);
  Xt[((size_t)s * BB + b) * DD + d] = (f16)q[i];
}

__global__ void cvt_kernel(const float* __restrict__ src, f16* __restrict__ dst, long n) {
  long i = (long)blockIdx.x * 256 + threadIdx.x;
  if (i < n) dst[i] = (f16)src[i];
}

__global__ void bias_sum_kernel(const float* __restrict__ bq, const float* __restrict__ bk,
                                const float* __restrict__ bv,
                                float* __restrict__ bqs, float* __restrict__ bks,
                                float* __restrict__ bvs) {
  int o = blockIdx.x * 256 + threadIdx.x;
  if (o >= DD) return;
  float a = 0.f, b = 0.f, c = 0.f;
  for (int m = 0; m < 5; ++m) {
    a += bq[m * DD + o];
    b += bk[m * DD + o];
    c += bv[m * DD + o];
  }
  bqs[o] = a; bks[o] = b; bvs[o] = c;
}

// zero padded seq slot (s = 511): q/k are [B,H,SP,DKV]; vt is [B,H,DKV,SP]
__global__ void zero_pad_kernel(f16* qb, f16* kb, f16* vtb) {
  int i = blockIdx.x * 256 + threadIdx.x;   // 256*64 elements
  if (i >= BB * HH * DKV) return;
  int bh = i >> 6, dk = i & 63;
  size_t idx = ((size_t)bh * SP + (SP - 1)) * DKV + dk;
  qb[idx] = (f16)0.f;
  kb[idx] = (f16)0.f;
  vtb[((size_t)bh * DKV + dk) * SP + (SP - 1)] = (f16)0.f;
}

// ---------------- QKV projection ----------------
// grid (64 col-tiles, 511 s). One wave per block. Rows of the tile = batch.
__global__ __launch_bounds__(32) void qkv_kernel(
    const f16* __restrict__ Xt, const f16* __restrict__ Wqh, const f16* __restrict__ Wkh,
    const f16* __restrict__ Wvh, const float* __restrict__ bqs, const float* __restrict__ bks,
    const float* __restrict__ bvs, f16* __restrict__ qb, f16* __restrict__ kb,
    f16* __restrict__ vtb) {
  const int nt = blockIdx.x;            // output 16-col tile (0..63)
  const int s  = blockIdx.y;            // 0..510
  const int lane = threadIdx.x;
  const int ln = lane & 15, hw = lane >> 4;
  const int strm = (s == 0) ? 0 : (s == 1) ? 1 : (s == 2) ? 2 : (3 + ((s - 3) & 1));

  const f16* arow = Xt + ((size_t)s * BB + ln) * DD;          // A row = batch ln
  const int n = nt * 16 + ln;                                  // output feature (B col)
  const size_t wof = (size_t)strm * DD * DD + (size_t)n * DD;
  const f16* wq = Wqh + wof;
  const f16* wk = Wkh + wof;
  const f16* wv = Wvh + wof;
  const int acb = hw * 8, bcb = hw * 16;

  v8f aq = {0,0,0,0,0,0,0,0}, ak = {0,0,0,0,0,0,0,0}, av = {0,0,0,0,0,0,0,0};
#pragma unroll 2
  for (int kk = 0; kk < DD; kk += 32) {
    v16h A  = frag_ld_a(arow, kk + acb);
    v16h Bq = frag_ld_b(wq, kk + bcb);
    v16h Bk = frag_ld_b(wk, kk + bcb);
    v16h Bv = frag_ld_b(wv, kk + bcb);
    aq = WMMA_F16(A, Bq, aq);
    ak = WMMA_F16(A, Bk, ak);
    av = WMMA_F16(A, Bv, av);
  }
  const int o = nt * 16 + ln;           // D-tile col n = ln
  const int h = o >> 6, dk = o & 63;
  const float bqv = bqs[o], bkv = bks[o], bvv = bvs[o];
#pragma unroll
  for (int r = 0; r < 8; ++r) {
    const int bidx = r + 8 * hw;        // D-tile row m = batch
    const int bh = bidx * HH + h;
    const size_t qi = ((size_t)bh * SP + s) * DKV + dk;
    qb[qi] = (f16)((aq[r] + bqv) * 0.125f);   // fold 1/sqrt(dk)
    kb[qi] = (f16)(ak[r] + bkv);
    // V stored transposed: [B,H,DKV,SP] so PV B-fragments are contiguous
    vtb[((size_t)bh * DKV + dk) * SP + s] = (f16)(av[r] + bvv);
  }
}

// ---------------- flash attention ----------------
// grid (B*H). 8 waves per block; whole K and V^T staged in LDS via TDM.
// LDS (f16 units): ldsK [SP][64] at 0, ldsVt [64][SP] at 32768, P [8][16][40] at 65536.
constexpr int LDS_K  = 0;
constexpr int LDS_VT = SP * DKV;            // 32768
constexpr int LDS_P  = LDS_VT + DKV * SP;   // 65536
constexpr int SMEM_F16 = LDS_P + 8 * 16 * 40;
constexpr size_t SMEM_BYTES = (size_t)SMEM_F16 * 2;   // 141312 B (< 320 KB WGP LDS)

__global__ __launch_bounds__(256) void attn_kernel(
    const f16* __restrict__ qb, const f16* __restrict__ kb, const f16* __restrict__ vtb,
    const float* __restrict__ bias, f16* __restrict__ AO) {
  extern __shared__ f16 smem[];
  const int bh = blockIdx.x;            // 0..255
  const int b = bh >> 4, h = bh & 15;
  const int tid = threadIdx.x;
  const int wid = tid >> 5, lane = tid & 31;
  const int ln = lane & 15, hw = lane >> 4;
  const int acb = hw * 8;

  const f16* qbase = qb + (size_t)bh * SP * DKV;
  const float* biasb = bias + (size_t)bh * SS * SS;

  // wave 0 DMAs K (512x64) and V^T (64x512) into LDS, then everyone syncs
  if (wid == 0) {
    const unsigned int base = __builtin_amdgcn_groupstaticsize();
    tdm_load_2d_f16(base + LDS_K * 2,  kb  + (size_t)bh * SP * DKV, DKV, SP);
    tdm_load_2d_f16(base + LDS_VT * 2, vtb + (size_t)bh * DKV * SP, SP, DKV);
    __builtin_amdgcn_s_wait_tensorcnt(0);
  }
  __syncthreads();

  f16* Pw = smem + LDS_P + wid * (16 * 40);   // per-wave private P staging

  for (int it = 0; it < 4; ++it) {
    const int qt = wid * 4 + it;        // 0..31
    const int qrow = qt * 16 + ln;      // row 511 is zero-padded
    const f16* qrp = qbase + (size_t)qrow * DKV;
    const v16h Aq0 = frag_ld_a(qrp, acb);
    const v16h Aq1 = frag_ld_a(qrp, 32 + acb);

    float rmax[8], rsum[8];
    v8f accO[4];
#pragma unroll
    for (int r = 0; r < 8; ++r) { rmax[r] = -1e30f; rsum[r] = 0.f; }
#pragma unroll
    for (int t = 0; t < 4; ++t) accO[t] = (v8f){0,0,0,0,0,0,0,0};

    for (int kt2 = 0; kt2 < SP / 32; ++kt2) {
#pragma unroll
      for (int t = 0; t < 2; ++t) {
        const int ks = (kt2 * 2 + t) * 16;
        const f16* krp = smem + LDS_K + (size_t)(ks + ln) * DKV;   // B col = key ks+ln
        v8f sc = {0,0,0,0,0,0,0,0};
        sc = WMMA_F16(Aq0, frag_ld_b(krp, hw * 16), sc);
        sc = WMMA_F16(Aq1, frag_ld_b(krp, 32 + hw * 16), sc);
        const int kn = ks + ln;
#pragma unroll
        for (int r = 0; r < 8; ++r) {
          const int qm = qt * 16 + r + 8 * hw;
          float v = sc[r];
          if (kn < SS && qm < SS) v += biasb[(size_t)qm * SS + kn];
          if (kn >= SS) v = -1e30f;
          // row reductions across the 16-lane half-group holding this row
          float mx = v;
          mx = fmaxf(mx, __shfl_xor(mx, 1, 32));
          mx = fmaxf(mx, __shfl_xor(mx, 2, 32));
          mx = fmaxf(mx, __shfl_xor(mx, 4, 32));
          mx = fmaxf(mx, __shfl_xor(mx, 8, 32));
          const float nm = fmaxf(rmax[r], mx);
          const float corr = __expf(rmax[r] - nm);
          rmax[r] = nm;
          const float p = __expf(v - nm);
          float ps = p;
          ps += __shfl_xor(ps, 1, 32);
          ps += __shfl_xor(ps, 2, 32);
          ps += __shfl_xor(ps, 4, 32);
          ps += __shfl_xor(ps, 8, 32);
          rsum[r] = rsum[r] * corr + ps;
#pragma unroll
          for (int dvt = 0; dvt < 4; ++dvt) accO[dvt][r] *= corr;
          Pw[(r + 8 * hw) * 40 + t * 16 + ln] = (f16)p;   // D-layout -> LDS
        }
      }
      // read P back as A fragment (16x32); per-wave DS ops are in-order
      const v16h Ap = frag_ld_a(Pw + (size_t)ln * 40, acb);
      const int kbase2 = kt2 * 32;
#pragma unroll
      for (int dvt = 0; dvt < 4; ++dvt) {
        // B[k][n] = Vt[dvt*16+n][kbase2+k] : contiguous from transposed V in LDS
        const f16* vrow = smem + LDS_VT + (size_t)(dvt * 16 + ln) * SP + kbase2;
        accO[dvt] = WMMA_F16(Ap, frag_ld_b(vrow, hw * 16), accO[dvt]);
      }
    }
    // epilogue: normalize, store AO[s][b][h*64+dv] as f16
#pragma unroll
    for (int r = 0; r < 8; ++r) {
      const int qm = qt * 16 + r + 8 * hw;
      if (qm < SS) {
        const float inv = 1.0f / rsum[r];
#pragma unroll
        for (int dvt = 0; dvt < 4; ++dvt) {
          const int c = h * DKV + dvt * 16 + ln;
          AO[((size_t)qm * BB + b) * DD + c] = (f16)(accO[dvt][r] * inv);
        }
      }
    }
  }
}

// ---------------- output projection ----------------
__global__ __launch_bounds__(32) void proj_kernel(
    const f16* __restrict__ AO, const f16* __restrict__ Wph, const float* __restrict__ bp,
    float* __restrict__ out) {
  const int nt = blockIdx.x;            // 0..63
  const int s  = blockIdx.y;            // 0..510
  const int lane = threadIdx.x;
  const int ln = lane & 15, hw = lane >> 4;
  const f16* arow = AO + ((size_t)s * BB + ln) * DD;
  const int d = nt * 16 + ln;
  const f16* wrow = Wph + (size_t)d * DD;   // out = AO @ Wp^T : B[k][n] = Wp[n][k]
  v8f acc = {0,0,0,0,0,0,0,0};
#pragma unroll 2
  for (int kk = 0; kk < DD; kk += 32) {
    v16h A  = frag_ld_a(arow, kk + hw * 8);
    v16h Bw = frag_ld_b(wrow, kk + hw * 16);
    acc = WMMA_F16(A, Bw, acc);
  }
  const float bpv = bp[d];
#pragma unroll
  for (int r = 0; r < 8; ++r) {
    const int bidx = r + 8 * hw;
    out[((size_t)bidx * SS + s) * DD + d] = acc[r] + bpv;
  }
}

// ---------------- host launcher ----------------
extern "C" void kernel_launch(void* const* d_in, const int* in_sizes, int n_in,
                              void* d_out, int out_size, void* d_ws, size_t ws_size,
                              hipStream_t stream) {
  const float* queries   = (const float*)d_in[0];
  const float* attn_bias = (const float*)d_in[4];
  const float* Wq = (const float*)d_in[5];
  const float* bq = (const float*)d_in[6];
  const float* Wk = (const float*)d_in[7];
  const float* bk = (const float*)d_in[8];
  const float* Wv = (const float*)d_in[9];
  const float* bv = (const float*)d_in[10];
  const float* Wp = (const float*)d_in[11];
  const float* bp = (const float*)d_in[12];
  float* out = (float*)d_out;

  char* ws = (char*)d_ws;
  size_t off = 0;
  auto take = [&](size_t bytes) -> char* {
    char* p = ws + off;
    off += (bytes + 255) & ~(size_t)255;
    return p;
  };
  f16*   Xt   = (f16*)take((size_t)SS * BB * DD * 2);          // 16.7 MB
  f16*   Wqh  = (f16*)take((size_t)5 * DD * DD * 2);           // 10.5 MB
  f16*   Wkh  = (f16*)take((size_t)5 * DD * DD * 2);
  f16*   Wvh  = (f16*)take((size_t)5 * DD * DD * 2);
  f16*   Wph  = (f16*)take((size_t)DD * DD * 2);               // 2.1 MB
  float* bqs  = (float*)take(DD * 4);
  float* bks  = (float*)take(DD * 4);
  float* bvs  = (float*)take(DD * 4);
  f16*   qbuf = (f16*)take((size_t)BB * HH * SP * DKV * 2);    // 16.8 MB x3
  f16*   kbuf = (f16*)take((size_t)BB * HH * SP * DKV * 2);
  f16*   vtbuf= (f16*)take((size_t)BB * HH * DKV * SP * 2);
  f16*   AO   = (f16*)take((size_t)SS * BB * DD * 2);          // 16.7 MB

  // stage 0: convert / pack
  {
    const long nx = (long)BB * SS * DD;
    pack_x_kernel<<<(nx + 255) / 256, 256, 0, stream>>>(queries, Xt);
    const long nw = (long)5 * DD * DD;
    cvt_kernel<<<(nw + 255) / 256, 256, 0, stream>>>(Wq, Wqh, nw);
    cvt_kernel<<<(nw + 255) / 256, 256, 0, stream>>>(Wk, Wkh, nw);
    cvt_kernel<<<(nw + 255) / 256, 256, 0, stream>>>(Wv, Wvh, nw);
    const long np = (long)DD * DD;
    cvt_kernel<<<(np + 255) / 256, 256, 0, stream>>>(Wp, Wph, np);
    bias_sum_kernel<<<(DD + 255) / 256, 256, 0, stream>>>(bq, bk, bv, bqs, bks, bvs);
    zero_pad_kernel<<<(BB * HH * DKV + 255) / 256, 256, 0, stream>>>(qbuf, kbuf, vtbuf);
  }
  // stage 1: QKV projection (WMMA)
  qkv_kernel<<<dim3(DD / 16, SS), 32, 0, stream>>>(Xt, Wqh, Wkh, Wvh, bqs, bks, bvs,
                                                   qbuf, kbuf, vtbuf);
  // stage 2: flash attention (WMMA + TDM-staged K/V in LDS)
  attn_kernel<<<dim3(BB * HH), 256, SMEM_BYTES, stream>>>(qbuf, kbuf, vtbuf, attn_bias, AO);
  // stage 3: output projection (WMMA)
  proj_kernel<<<dim3(DD / 16, SS), 32, 0, stream>>>(AO, Wph, bp, out);
}
